// GraphModule_30820685316772
// MI455X (gfx1250) — matrix operand order repaired and checked by reading, not compiled, not run on hardware
//
#include <hip/hip_runtime.h>
#include <hip/hip_bf16.h>
#include <stdint.h>

typedef __bf16 bf16_t;
typedef __attribute__((ext_vector_type(16))) __bf16 bf16x16;
typedef __attribute__((ext_vector_type(8))) float f32x8;

#define D_MODEL 768
#define D_QKV   2304
#define D_FF    3072
#define N_HEAD  12
#define HEAD_DIM 64
#define SEQ     1024
#define BATCH   4
#define ROWS    4096   // BATCH*SEQ
#define N_LAYERS 5

// ---------------------------------------------------------------- helpers
__device__ __forceinline__ bf16_t f2bf(float f) {
  uint32_t u = __builtin_bit_cast(uint32_t, f);
  u += 0x7FFFu + ((u >> 16) & 1u);           // round-to-nearest-even
  uint16_t h = (uint16_t)(u >> 16);
  return __builtin_bit_cast(bf16_t, h);
}

__device__ __forceinline__ float geluf(float x) {
  const float c = 0.7978845608028654f;       // sqrt(2/pi)
  float t = tanhf(c * (x + 0.044715f * x * x * x));
  return 0.5f * x * (1.0f + t);
}

__device__ __forceinline__ f32x8 wmma_bf16(bf16x16 a, bf16x16 b, f32x8 c) {
  // D = A(16x32) * B(32x16) + C, fp32 accum
  return __builtin_amdgcn_wmma_f32_16x16x32_bf16(false, a, false, b,
                                                 (short)0, c, false, false);
}

// A-side 16x32 bf16 fragment: K index held in VGPR v (pair base), lane half hi
__device__ __forceinline__ int kpatA(int v, int hi) {
  return 2 * v + (v >= 4 ? 8 : 0) + hi * 8;
}
// B-side 32x16 bf16 fragment: K index for VGPR v, lane half hi
__device__ __forceinline__ int kpatB(int v, int hi) {
  return 2 * v + (v >= 4 ? 8 : 0) + hi * 8;
}

// ---------------------------------------------------------------- embed
__global__ __launch_bounds__(256)
void embed_kernel(const int* __restrict__ ids, const float* __restrict__ tok,
                  const float* __restrict__ pos, float* __restrict__ h,
                  bf16_t* __restrict__ hb) {
  const int row = blockIdx.x;            // b*SEQ + s
  const int s = row & (SEQ - 1);
  const int id = ids[row];
#pragma unroll
  for (int t = 0; t < 3; ++t) {
    const int c = threadIdx.x + t * 256;
    float v = tok[(size_t)id * D_MODEL + c] + pos[(size_t)s * D_MODEL + c];
    h[(size_t)row * D_MODEL + c] = v;
    hb[(size_t)row * D_MODEL + c] = f2bf(v);
  }
}

// ---------------------------------------------------------------- convert
__global__ __launch_bounds__(256)
void f32_to_bf16_kernel(const float* __restrict__ src, bf16_t* __restrict__ dst,
                        int n) {
  int i = blockIdx.x * 256 + threadIdx.x;
  if (i < n) dst[i] = f2bf(src[i]);
}

// ---------------------------------------------------------------- GEMM
// C[M,N] = A[M,K]@B[K,N] + bias (+resid), optional GELU.
// Block: 256 threads (8 waves), tile 128x128, K-step 32, double-buffered LDS.
#define BM 128
#define BN 128
#define BK 32

__device__ __forceinline__ void load_tile_regs(const bf16_t* __restrict__ A,
                                               const bf16_t* __restrict__ B,
                                               int N, int K, int mBase,
                                               int nBase, int k0, int tid,
                                               uint32_t aR[8], uint32_t bR[8]) {
#pragma unroll
  for (int t = 0; t < 8; ++t) {
    const int l = tid + t * 256;
    aR[t] = *(const uint32_t*)&A[(size_t)(mBase + (l >> 4)) * K + k0 +
                                 (l & 15) * 2];
  }
#pragma unroll
  for (int t = 0; t < 8; ++t) {
    const int l = tid + t * 256;
    bR[t] = *(const uint32_t*)&B[(size_t)(k0 + (l >> 6)) * N + nBase +
                                 (l & 63) * 2];
  }
}

__device__ __forceinline__ void store_tile_lds(bf16_t (*As)[BK],
                                               bf16_t (*Bs)[BK], int tid,
                                               const uint32_t aR[8],
                                               const uint32_t bR[8]) {
#pragma unroll
  for (int t = 0; t < 8; ++t) {
    const int l = tid + t * 256;
    *(uint32_t*)&As[l >> 4][(l & 15) * 2] = aR[t];
  }
#pragma unroll
  for (int t = 0; t < 8; ++t) {
    const int l = tid + t * 256;
    const int np = (l & 63) * 2;
    Bs[np][l >> 6]     = ((const bf16_t*)&bR[t])[0];
    Bs[np + 1][l >> 6] = ((const bf16_t*)&bR[t])[1];
  }
}

template <bool HAS_RESID, bool GELU, bool OUT_F32>
__global__ __launch_bounds__(256)
void gemm_bf16_t(const bf16_t* __restrict__ A, const bf16_t* __restrict__ B,
                 const float* __restrict__ bias,
                 const float* __restrict__ resid, float* __restrict__ Cf,
                 bf16_t* __restrict__ Cb, int M, int N, int K) {
  __shared__ bf16_t As[2][BM][BK];   // 16 KB (ping-pong)
  __shared__ bf16_t Bs[2][BN][BK];   // 16 KB (transposed tiles: [n][k])

  const int tid = threadIdx.x;
  const int lane = tid & 31;
  const int wave = tid >> 5;
  const int hi = (lane >> 4) & 1;
  const int l16 = lane & 15;
  const int wm = wave >> 1;       // 4 waves along M
  const int wn = wave & 1;        // 2 waves along N
  const int mBase = blockIdx.y * BM;
  const int nBase = blockIdx.x * BN;

  f32x8 acc[2][4] = {};

  uint32_t aR[8], bR[8];
  load_tile_regs(A, B, N, K, mBase, nBase, 0, tid, aR, bR);
  store_tile_lds(As[0], Bs[0], tid, aR, bR);

  int cur = 0;
  for (int k0 = 0; k0 < K; k0 += BK) {
    __syncthreads();
    const bool hasNext = (k0 + BK) < K;
    if (hasNext) {
      load_tile_regs(A, B, N, K, mBase, nBase, k0 + BK, tid, aR, bR);
      if (k0 + 2 * BK < K) {   // stream tile k0+2*BK toward the WGP
        __builtin_prefetch(&A[(size_t)(mBase + (tid >> 1)) * K + k0 + 2 * BK],
                           0, 3);
        __builtin_prefetch(&B[(size_t)(k0 + 2 * BK + (tid >> 3)) * N + nBase],
                           0, 3);
      }
    }

    bf16x16 af[2], bfr[4];
#pragma unroll
    for (int i = 0; i < 2; ++i) {
      const int m = wm * 32 + i * 16 + l16;
#pragma unroll
      for (int v = 0; v < 8; ++v) {
        const int kb = kpatA(v, hi);
        af[i][2 * v]     = As[cur][m][kb];
        af[i][2 * v + 1] = As[cur][m][kb + 1];
      }
    }
#pragma unroll
    for (int j = 0; j < 4; ++j) {
      const int n = wn * 64 + j * 16 + l16;
#pragma unroll
      for (int v = 0; v < 8; ++v) {
        const int kb = kpatB(v, hi);
        bfr[j][2 * v]     = Bs[cur][n][kb];
        bfr[j][2 * v + 1] = Bs[cur][n][kb + 1];
      }
    }
#pragma unroll
    for (int i = 0; i < 2; ++i)
#pragma unroll
      for (int j = 0; j < 4; ++j)
        acc[i][j] = wmma_bf16(af[i], bfr[j], acc[i][j]);

    if (hasNext) store_tile_lds(As[cur ^ 1], Bs[cur ^ 1], tid, aR, bR);
    cur ^= 1;
  }

  // straight-line epilogue (all policy decisions are compile-time)
#pragma unroll
  for (int i = 0; i < 2; ++i) {
#pragma unroll
    for (int j = 0; j < 4; ++j) {
      const int n = nBase + wn * 64 + j * 16 + l16;
      const float bv = bias[n];
#pragma unroll
      for (int r = 0; r < 8; ++r) {
        const int m = mBase + wm * 32 + i * 16 + r + hi * 8;
        float x = acc[i][j][r] + bv;
        if constexpr (HAS_RESID) x += resid[(size_t)m * N + n];
        if constexpr (GELU) x = geluf(x);
        if constexpr (OUT_F32)
          Cf[(size_t)m * N + n] = x;
        else
          Cb[(size_t)m * N + n] = f2bf(x);
      }
    }
  }
}

// ---------------------------------------------------------------- attention
// Flash-style causal attention over fused qkv (bf16, [B,S,3D], head-major).
// Grid: (SEQ/64, B*H). Block: 128 threads = 4 waves; each wave owns 16 query
// rows, loops over 64-key blocks with streaming softmax; output written as
// bf16 in [B,S,D] layout (head-major) directly usable by the proj GEMM.
__global__ __launch_bounds__(128)
void attn_kernel(const bf16_t* __restrict__ qkv, const float* __restrict__ amask,
                 bf16_t* __restrict__ aout) {
  __shared__ bf16_t Pb[4][16][64];  // per-wave P staging, 8 KB

  const int qblk = blockIdx.x;      // 0..15
  const int bh = blockIdx.y;        // 0..47
  const int b = bh / N_HEAD;
  const int head = bh % N_HEAD;
  const int tid = threadIdx.x;
  const int lane = tid & 31;
  const int wave = tid >> 5;
  const int hi = (lane >> 4) & 1;
  const int l16 = lane & 15;
  const int qBase = qblk * 64;

  const bf16_t* Qp = qkv + head * HEAD_DIM;
  const bf16_t* Kp = qkv + D_MODEL + head * HEAD_DIM;
  const bf16_t* Vp = qkv + 2 * D_MODEL + head * HEAD_DIM;

  // Q A-fragments for this wave's 16 rows (HD=64 -> 2 K-steps), kept in regs
  bf16x16 qa[2];
  {
    const int qr = qBase + wave * 16 + l16;
    const size_t base = (size_t)(b * SEQ + qr) * D_QKV;
#pragma unroll
    for (int ks = 0; ks < 2; ++ks) {
#pragma unroll
      for (int v = 0; v < 8; ++v) {
        const int kd = ks * 32 + kpatA(v, hi);
        qa[ks][2 * v]     = Qp[base + kd];
        qa[ks][2 * v + 1] = Qp[base + kd + 1];
      }
    }
  }

  f32x8 oacc[4] = {};
  float rmax[8], rsum[8];
#pragma unroll
  for (int r = 0; r < 8; ++r) { rmax[r] = -1e30f; rsum[r] = 0.0f; }

  for (int kb = 0; kb <= qblk; ++kb) {
    const int keyBase = kb * 64;

    // ---- load ALL K fragments first (distinct regs -> one load clause,
    //      latency overlapped with the WMMA stream) ----
    bf16x16 kf[4][2];
#pragma unroll
    for (int j = 0; j < 4; ++j) {
      const int key = keyBase + j * 16 + l16;
      const size_t base = (size_t)(b * SEQ + key) * D_QKV;
#pragma unroll
      for (int ks = 0; ks < 2; ++ks) {
#pragma unroll
        for (int v = 0; v < 8; ++v) {
          const int kd = ks * 32 + kpatB(v, hi);
          kf[j][ks][2 * v]     = Kp[base + kd];
          kf[j][ks][2 * v + 1] = Kp[base + kd + 1];
        }
      }
    }

    // ---- S = Q @ K^T (16 rows x 64 keys) ----
    f32x8 sacc[4] = {};
#pragma unroll
    for (int j = 0; j < 4; ++j) {
#pragma unroll
      for (int ks = 0; ks < 2; ++ks)
        sacc[j] = wmma_bf16(qa[ks], kf[j][ks], sacc[j]);
    }

    // ---- scale + causal mask (-1e4 per GPT-1) + ext mask, streaming softmax
    float pv[4][8];
#pragma unroll
    for (int r = 0; r < 8; ++r) {
      const int qr = qBase + wave * 16 + r + hi * 8;
      float rowm = -1e30f;
#pragma unroll
      for (int j = 0; j < 4; ++j) {
        const int key = keyBase + j * 16 + l16;
        float x = (key <= qr) ? sacc[j][r] * 0.125f : -10000.0f;
        x += (1.0f - amask[b * SEQ + key]) * -1e9f;
        pv[j][r] = x;
        rowm = fmaxf(rowm, x);
      }
#pragma unroll
      for (int off = 1; off < 16; off <<= 1)
        rowm = fmaxf(rowm, __shfl_xor(rowm, off, 32));
      const float nmax = fmaxf(rmax[r], rowm);
      const float corr = __expf(rmax[r] - nmax);
      float rs = 0.0f;
#pragma unroll
      for (int j = 0; j < 4; ++j) {
        const float e = __expf(pv[j][r] - nmax);
        pv[j][r] = e;
        rs += e;
      }
#pragma unroll
      for (int off = 1; off < 16; off <<= 1)
        rs += __shfl_xor(rs, off, 32);
      rsum[r] = rsum[r] * corr + rs;
      rmax[r] = nmax;
#pragma unroll
      for (int jo = 0; jo < 4; ++jo) oacc[jo][r] *= corr;
    }

    // ---- stage P through LDS to re-layout C-frag -> A-frag ----
    __syncthreads();
#pragma unroll
    for (int r = 0; r < 8; ++r) {
      const int lr = r + hi * 8;
#pragma unroll
      for (int j = 0; j < 4; ++j)
        Pb[wave][lr][j * 16 + l16] = f2bf(pv[j][r]);
    }
    __syncthreads();

    bf16x16 pa[2];
#pragma unroll
    for (int ks = 0; ks < 2; ++ks) {
#pragma unroll
      for (int v = 0; v < 8; ++v) {
        const int kk = ks * 32 + kpatA(v, hi);
        pa[ks][2 * v]     = Pb[wave][l16][kk];
        pa[ks][2 * v + 1] = Pb[wave][l16][kk + 1];
      }
    }

    // ---- load ALL V fragments, then O += P @ V ----
    bf16x16 vf[4][2];
#pragma unroll
    for (int jo = 0; jo < 4; ++jo) {
#pragma unroll
      for (int ks = 0; ks < 2; ++ks) {
#pragma unroll
        for (int v = 0; v < 8; ++v) {
          const int kk = ks * 32 + kpatB(v, hi);
          const size_t base0 = (size_t)(b * SEQ + keyBase + kk) * D_QKV;
          vf[jo][ks][2 * v]     = Vp[base0 + jo * 16 + l16];
          vf[jo][ks][2 * v + 1] = Vp[base0 + D_QKV + jo * 16 + l16];
        }
      }
    }
#pragma unroll
    for (int jo = 0; jo < 4; ++jo) {
#pragma unroll
      for (int ks = 0; ks < 2; ++ks)
        oacc[jo] = wmma_bf16(pa[ks], vf[jo][ks], oacc[jo]);
    }
  }

  // ---- normalize and write [B,S,D] bf16 (head-major cols) ----
#pragma unroll
  for (int jo = 0; jo < 4; ++jo) {
#pragma unroll
    for (int r = 0; r < 8; ++r) {
      const int qr = qBase + wave * 16 + r + hi * 8;
      const float o = oacc[jo][r] / rsum[r];
      aout[(size_t)(b * SEQ + qr) * D_MODEL + head * HEAD_DIM + jo * 16 + l16] =
          f2bf(o);
    }
  }
}

// ---------------------------------------------------------------- layernorm
__global__ __launch_bounds__(256)
void layernorm_kernel(const float* __restrict__ in, const float* __restrict__ gw,
                      const float* __restrict__ gb, float* __restrict__ outf,
                      bf16_t* __restrict__ outb) {
  __shared__ float red[8];
  const int row = blockIdx.x;
  const float* x = in + (size_t)row * D_MODEL;
  const float v0 = x[threadIdx.x];
  const float v1 = x[threadIdx.x + 256];
  const float v2 = x[threadIdx.x + 512];

  float s = v0 + v1 + v2;
#pragma unroll
  for (int off = 16; off > 0; off >>= 1) s += __shfl_xor(s, off, 32);
  if ((threadIdx.x & 31) == 0) red[threadIdx.x >> 5] = s;
  __syncthreads();
  float tot = 0.0f;
#pragma unroll
  for (int i = 0; i < 8; ++i) tot += red[i];
  const float mean = tot * (1.0f / D_MODEL);
  __syncthreads();

  const float d0 = v0 - mean, d1 = v1 - mean, d2 = v2 - mean;
  float s2 = d0 * d0 + d1 * d1 + d2 * d2;
#pragma unroll
  for (int off = 16; off > 0; off >>= 1) s2 += __shfl_xor(s2, off, 32);
  if ((threadIdx.x & 31) == 0) red[threadIdx.x >> 5] = s2;
  __syncthreads();
  float tot2 = 0.0f;
#pragma unroll
  for (int i = 0; i < 8; ++i) tot2 += red[i];
  const float rstd = rsqrtf(tot2 * (1.0f / D_MODEL) + 1e-5f);

  const float dd[3] = {d0, d1, d2};
#pragma unroll
  for (int t = 0; t < 3; ++t) {
    const int c = threadIdx.x + t * 256;
    const float y = dd[t] * rstd * gw[c] + gb[c];
    outf[(size_t)row * D_MODEL + c] = y;
    outb[(size_t)row * D_MODEL + c] = f2bf(y);
  }
}

// ---------------------------------------------------------------- launch
extern "C" void kernel_launch(void* const* d_in, const int* in_sizes, int n_in,
                              void* d_out, int out_size, void* d_ws,
                              size_t ws_size, hipStream_t stream) {
  (void)in_sizes; (void)n_in; (void)out_size; (void)ws_size;
  const int*   input_ids = (const int*)d_in[0];
  // d_in[1] = position_ids (== arange(S)); index math uses s directly
  const float* amask     = (const float*)d_in[2];
  const float* tok_emb   = (const float*)d_in[3];
  const float* pos_emb   = (const float*)d_in[4];
  const float* attn_w    = (const float*)d_in[5];
  const float* attn_b    = (const float*)d_in[6];
  const float* proj_w    = (const float*)d_in[7];
  const float* proj_b    = (const float*)d_in[8];
  const float* ln1w      = (const float*)d_in[9];
  const float* ln1b      = (const float*)d_in[10];
  const float* fc_w      = (const float*)d_in[11];
  const float* fc_b      = (const float*)d_in[12];
  const float* mlp_w     = (const float*)d_in[13];
  const float* mlp_b     = (const float*)d_in[14];
  const float* ln2w      = (const float*)d_in[15];
  const float* ln2b      = (const float*)d_in[16];
  float* out = (float*)d_out;

  char* p = (char*)d_ws;
  auto take = [&](size_t bytes) {
    char* r = p;
    p += (bytes + 255) & ~(size_t)255;
    return r;
  };
  bf16_t* wattn = (bf16_t*)take((size_t)D_MODEL * D_QKV * 2);
  bf16_t* wproj = (bf16_t*)take((size_t)D_MODEL * D_MODEL * 2);
  bf16_t* wfc   = (bf16_t*)take((size_t)D_MODEL * D_FF * 2);
  bf16_t* wmlp  = (bf16_t*)take((size_t)D_FF * D_MODEL * 2);
  float*  hbuf  = (float*)take((size_t)ROWS * D_MODEL * 4);
  bf16_t* hb    = (bf16_t*)take((size_t)ROWS * D_MODEL * 2);
  bf16_t* qkvb  = (bf16_t*)take((size_t)ROWS * D_QKV * 2);
  bf16_t* ab    = (bf16_t*)take((size_t)ROWS * D_MODEL * 2);
  float*  sum1  = (float*)take((size_t)ROWS * D_MODEL * 4);
  float*  nbuf  = (float*)take((size_t)ROWS * D_MODEL * 4);
  bf16_t* nb    = (bf16_t*)take((size_t)ROWS * D_MODEL * 2);
  bf16_t* fcb   = (bf16_t*)take((size_t)ROWS * D_FF * 2);

  embed_kernel<<<ROWS, 256, 0, stream>>>(input_ids, tok_emb, pos_emb, hbuf, hb);

  for (int i = 0; i < N_LAYERS; ++i) {
    const int nA = D_MODEL * D_QKV;
    const int nP = D_MODEL * D_MODEL;
    const int nF = D_MODEL * D_FF;
    f32_to_bf16_kernel<<<(nA + 255) / 256, 256, 0, stream>>>(
        attn_w + (size_t)i * nA, wattn, nA);
    f32_to_bf16_kernel<<<(nP + 255) / 256, 256, 0, stream>>>(
        proj_w + (size_t)i * nP, wproj, nP);
    f32_to_bf16_kernel<<<(nF + 255) / 256, 256, 0, stream>>>(
        fc_w + (size_t)i * nF, wfc, nF);
    f32_to_bf16_kernel<<<(nF + 255) / 256, 256, 0, stream>>>(
        mlp_w + (size_t)i * nF, wmlp, nF);

    // qkv = h @ Wqkv + b  (bf16 out, fused head-major q|k|v rows)
    gemm_bf16_t<false, false, false>
        <<<dim3(D_QKV / BN, ROWS / BM), 256, 0, stream>>>(
            hb, wattn, attn_b + (size_t)i * D_QKV, nullptr, nullptr, qkvb,
            ROWS, D_QKV, D_MODEL);

    attn_kernel<<<dim3(SEQ / 64, BATCH * N_HEAD), 128, 0, stream>>>(qkvb, amask,
                                                                    ab);

    // sum1 = h + (a @ Wproj + b)   (fp32 out)
    gemm_bf16_t<true, false, true>
        <<<dim3(D_MODEL / BN, ROWS / BM), 256, 0, stream>>>(
            ab, wproj, proj_b + (size_t)i * D_MODEL, hbuf, sum1, nullptr,
            ROWS, D_MODEL, D_MODEL);

    layernorm_kernel<<<ROWS, 256, 0, stream>>>(
        sum1, ln1w + (size_t)i * D_MODEL, ln1b + (size_t)i * D_MODEL, nbuf, nb);

    // fcb = gelu(n @ Wfc + b)  (bf16 out)
    gemm_bf16_t<false, true, false>
        <<<dim3(D_FF / BN, ROWS / BM), 256, 0, stream>>>(
            nb, wfc, fc_b + (size_t)i * D_FF, nullptr, nullptr, fcb,
            ROWS, D_FF, D_MODEL);

    // sum1 = n + (m @ Wmlp + b)   (fp32 out)
    gemm_bf16_t<true, false, true>
        <<<dim3(D_MODEL / BN, ROWS / BM), 256, 0, stream>>>(
            fcb, wmlp, mlp_b + (size_t)i * D_MODEL, nbuf, sum1, nullptr,
            ROWS, D_MODEL, D_FF);

    float* outf = (i == N_LAYERS - 1) ? out : hbuf;
    layernorm_kernel<<<ROWS, 256, 0, stream>>>(
        sum1, ln2w + (size_t)i * D_MODEL, ln2b + (size_t)i * D_MODEL, outf, hb);
  }
}
